// MSA_att_10926396801701
// MI455X (gfx1250) — matrix operand order repaired and checked by reading, not compiled, not run on hardware
//
#include <hip/hip_runtime.h>

typedef __attribute__((ext_vector_type(16))) _Float16 v16h;
typedef __attribute__((ext_vector_type(8)))  float    v8f;

#define USE_ASYNC 1   // gfx1250 async global->LDS staging (ASYNCcnt)

// ---------------------------------------------------------------------------
// WMMA fragment loaders (CDNA5 ISA 7.12.2, wave32, 16x16x32 f16)
// A tile in LDS: row-major [m][k], padded stride LDT halves.
// B tile in LDS: stored transposed Bt[n][k] so each lane reads 16 contiguous k.
// LDT = 40 halves = 80 B (multiple of 16B) -> all frag reads are aligned b128.
// ---------------------------------------------------------------------------
#define LDT 40

__device__ inline v16h frag_a(const _Float16* r, int hv) {
  v16h a;
#pragma unroll
  for (int j = 0; j < 4; ++j) {
    int k0 = hv * 16 + 2 * j;
    a[2 * j]     = r[k0];
    a[2 * j + 1] = r[k0 + 1];
    int k1 = k0 + 8;
    a[8 + 2 * j]     = r[k1];
    a[8 + 2 * j + 1] = r[k1 + 1];
  }
  return a;
}

__device__ inline v16h frag_b(const _Float16* r) {
  v16h b;
#pragma unroll
  for (int e = 0; e < 16; ++e) b[e] = r[e];
  return b;
}

#if USE_ASYNC
__device__ inline void async_copy16(const _Float16* g, _Float16* l) {
  unsigned lds = (unsigned)(size_t)l;                       // LDS offset (addr[31:0])
  unsigned long long ga = (unsigned long long)(size_t)g;
  asm volatile("global_load_async_to_lds_b128 %0, %1, off"
               :: "v"(lds), "v"(ga) : "memory");
}
__device__ inline void async_wait0() {
  asm volatile("s_wait_asynccnt 0" ::: "memory");
}
#else
__device__ inline void async_copy16(const _Float16* g, _Float16* l) {
  *(float4*)l = *(const float4*)g;
}
__device__ inline void async_wait0() {}
#endif

// ---------------------------------------------------------------------------
// Generic batched WMMA GEMM, double-buffered async pipeline.
//   C[M,N] (+C16) = op(alpha * A[M,K] x B + bias) + resid
//   transB=1: B given as [N,K] row-major (direct async-copy staging)
//   transB=0: B given as [K,N] row-major (transposing staging)
//   grid.z = Z batches; b = z / H, h = z % H; per-operand (b,h) strides.
// Block tile 256x64, wave tile 32x64, K-step 32.
// Requires M%256==0, N%64==0, K%32==0 (true for every call in this model).
// ---------------------------------------------------------------------------
__global__ __launch_bounds__(256)
void k_gemm(const _Float16* A, const _Float16* Bm, float* C, _Float16* C16,
            const float* bias, const float* resid,
            int K, long lda, long ldb, long ldc,
            long Asb, long Ash, long Bsb, long Bsh, long Csb, long Csh,
            int H, int transB, int op, float alpha) {
  __shared__ _Float16 As[2][256 * LDT];   // 2 x 20480 B
  __shared__ _Float16 Bs[2][64 * LDT];    // 2 x  5120 B

  int z = blockIdx.z;
  int bb = z / H, hh = z % H;
  A  += (long)bb * Asb + (long)hh * Ash;
  Bm += (long)bb * Bsb + (long)hh * Bsh;
  long coff = (long)bb * Csb + (long)hh * Csh;

  int m0 = blockIdx.y * 256;
  int n0 = blockIdx.x * 64;
  int t = threadIdx.x;
  int wave = t >> 5, lane = t & 31;
  int mlane = lane & 15, hv = lane >> 4;

  // stage tile at K-offset kk into buffer buf
  auto stage = [&](int buf, int kk) {
    _Float16* Ad = As[buf];
    _Float16* Bd = Bs[buf];
    // A: 256x32 halves = 1024 x 16B chunks, 4 per thread (async)
#pragma unroll
    for (int i = 0; i < 4; ++i) {
      int slot = t + i * 256;            // 0..1023
      int m  = slot >> 2;
      int kd = (slot & 3) << 3;          // 0,8,16,24
      async_copy16(A + (long)(m0 + m) * lda + kk + kd, &Ad[m * LDT + kd]);
    }
    if (transB) {
      // B: 64x32 halves = 256 x 16B chunks, 1 per thread (async)
      int n  = t >> 2;
      int kd = (t & 3) << 3;
      async_copy16(Bm + (long)(n0 + n) * ldb + kk + kd, &Bd[n * LDT + kd]);
    } else {
      // transpose: 16B vector load along n, scatter b16 into Bt[n][k]
      int k  = t >> 3;                   // 0..31
      int nn = (t & 7) << 3;             // 0..56
      union { float4 f; _Float16 hsm[8]; } u;
      u.f = *(const float4*)(Bm + (long)(kk + k) * ldb + n0 + nn);
#pragma unroll
      for (int j = 0; j < 8; ++j) Bd[(nn + j) * LDT + k] = u.hsm[j];
    }
  };

  v8f zero = {0.f, 0.f, 0.f, 0.f, 0.f, 0.f, 0.f, 0.f};
  v8f acc[2][4];
#pragma unroll
  for (int i = 0; i < 2; ++i)
#pragma unroll
    for (int j = 0; j < 4; ++j) acc[i][j] = zero;

  // prologue: fill buffer 0
  stage(0, 0);
  async_wait0();
  __syncthreads();

  int cur = 0;
  for (int k0 = 0; k0 < K; k0 += 32) {
    // prefetch next tile into the alternate buffer while computing this one
    if (k0 + 32 < K) stage(cur ^ 1, k0 + 32);

    int mrow = wave * 32;
    const _Float16* Ac = As[cur];
    const _Float16* Bc = Bs[cur];
    v16h a0 = frag_a(&Ac[(mrow + mlane) * LDT], hv);
    v16h a1 = frag_a(&Ac[(mrow + 16 + mlane) * LDT], hv);
#pragma unroll
    for (int nt = 0; nt < 4; ++nt) {
      v16h bf = frag_b(&Bc[(nt * 16 + mlane) * LDT + hv * 16]);
      acc[0][nt] = __builtin_amdgcn_wmma_f32_16x16x32_f16(
          false, a0, false, bf, (short)0, acc[0][nt], false, false);
      acc[1][nt] = __builtin_amdgcn_wmma_f32_16x16x32_f16(
          false, a1, false, bf, (short)0, acc[1][nt], false, false);
    }
    async_wait0();       // next tile's async transfers (this wave) complete
    __syncthreads();     // publish across waves; cur reads are done (dscnt-waited)
    cur ^= 1;
  }

  // ---- epilogue: lane holds col n = nt*16 + mlane, rows v + 8*hv (per vgpr)
#pragma unroll
  for (int mt = 0; mt < 2; ++mt) {
#pragma unroll
    for (int nt = 0; nt < 4; ++nt) {
      int n = n0 + nt * 16 + mlane;
#pragma unroll
      for (int v = 0; v < 8; ++v) {
        int m = m0 + wave * 32 + mt * 16 + v + 8 * hv;
        float c = acc[mt][nt][v] * alpha;
        if (bias) c += bias[n];
        if (op == 1) c = fmaxf(c, 0.f);
        else if (op == 2) c = tanhf(c);
        long idx = coff + (long)m * ldc + n;
        if (resid) c += resid[idx];
        if (C)   C[idx] = c;
        if (C16) C16[idx] = (_Float16)c;
      }
    }
  }
}

// ---------------------------------------------------------------------------
// Elementwise / reduction helper kernels
// ---------------------------------------------------------------------------
__global__ void k_f2h(const float* in, _Float16* out, long n) {
  long i = (long)blockIdx.x * 256 + threadIdx.x;
  if (i < n) out[i] = (_Float16)in[i];
}

__global__ void k_layernorm(const float* x, const float* g, const float* bta,
                            _Float16* out, int dim) {
  long row = blockIdx.x;
  const float* xr = x + row * (long)dim;
  __shared__ float red[256];
  int tid = threadIdx.x;
  float s = 0.f;
  for (int c = tid; c < dim; c += 256) s += xr[c];
  red[tid] = s; __syncthreads();
  for (int o = 128; o > 0; o >>= 1) { if (tid < o) red[tid] += red[tid + o]; __syncthreads(); }
  float mu = red[0] / dim; __syncthreads();
  float v = 0.f;
  for (int c = tid; c < dim; c += 256) { float d = xr[c] - mu; v += d * d; }
  red[tid] = v; __syncthreads();
  for (int o = 128; o > 0; o >>= 1) { if (tid < o) red[tid] += red[tid + o]; __syncthreads(); }
  float rs = rsqrtf(red[0] / dim + 1e-5f); __syncthreads();
  for (int c = tid; c < dim; c += 256)
    out[row * (long)dim + c] = (_Float16)((xr[c] - mu) * rs * g[c] + bta[c]);
}

__global__ void k_row_softmax(_Float16* s, float* out32, int cols) {
  long row = blockIdx.x;
  _Float16* r = s + row * (long)cols;
  __shared__ float red[256];
  int tid = threadIdx.x;
  float m = -1e30f;
  for (int c = tid; c < cols; c += 256) m = fmaxf(m, (float)r[c]);
  red[tid] = m; __syncthreads();
  for (int o = 128; o > 0; o >>= 1) { if (tid < o) red[tid] = fmaxf(red[tid], red[tid + o]); __syncthreads(); }
  m = red[0]; __syncthreads();
  float sum = 0.f;
  for (int c = tid; c < cols; c += 256) sum += expf((float)r[c] - m);
  red[tid] = sum; __syncthreads();
  for (int o = 128; o > 0; o >>= 1) { if (tid < o) red[tid] += red[tid + o]; __syncthreads(); }
  float inv = 1.f / red[0]; __syncthreads();
  for (int c = tid; c < cols; c += 256) {
    float p = expf((float)r[c] - m) * inv;
    r[c] = (_Float16)p;
    if (out32) out32[row * (long)cols + c] = p;
  }
}

// landmark means: out[bh,l,d] = mean_{s<64} qkv[(b*16384 + l*64+s)*1536 + colofs + h*64 + d]
__global__ void k_landmark(const _Float16* qkv, _Float16* out, int colofs) {
  long idx = (long)blockIdx.x * 256 + threadIdx.x;  // 16*256*64
  int d = idx & 63; long r = idx >> 6;
  int l = (int)(r & 255); int bh = (int)(r >> 8);
  int b = bh >> 3, h2 = bh & 7;
  const _Float16* base = qkv + ((long)(b * 16384 + l * 64)) * 1536 + colofs + h2 * 64 + d;
  float s = 0.f;
  for (int k = 0; k < 64; ++k) s += (float)base[(long)k * 1536];
  out[idx] = (_Float16)(s * (1.f / 64.f));
}

__global__ void k_abssum_lastdim(const float* x, float* out) {
  long r = (long)blockIdx.x * 256 + threadIdx.x;      // 4096 rows of 256
  const float* p = x + r * 256;
  float s = 0.f;
  for (int j = 0; j < 256; ++j) s += fabsf(p[j]);
  out[r] = s;
}

__global__ void k_abssum_dim2(const float* x, float* out) {
  long r = (long)blockIdx.x * 256 + threadIdx.x;      // bh*256 + j
  int j = (int)(r & 255); int bh = (int)(r >> 8);
  const float* p = x + (long)bh * 65536 + j;
  float s = 0.f;
  for (int i = 0; i < 256; ++i) s += fabsf(p[i * 256]);
  out[r] = s;
}

__global__ void k_maxreduce(const float* in, int n, float* out) {
  __shared__ float red[256];
  int tid = threadIdx.x;
  float m = -1e30f;
  for (int i = tid; i < n; i += 256) m = fmaxf(m, in[i]);
  red[tid] = m; __syncthreads();
  for (int o = 128; o > 0; o >>= 1) { if (tid < o) red[tid] = fmaxf(red[tid], red[tid + o]); __syncthreads(); }
  if (tid == 0) out[0] = red[0];
}

// z0 = x^T / (col*row)
__global__ void k_z0(const float* x2, const float* col, const float* row, _Float16* z) {
  long idx = (long)blockIdx.x * 256 + threadIdx.x;    // 16*65536
  int j = (int)(idx & 255); long r = idx >> 8;
  int i = (int)(r & 255); int bh = (int)(r >> 8);
  float inv = 1.f / (col[0] * row[0]);
  z[(long)bh * 65536 + i * 256 + j] = (_Float16)(x2[(long)bh * 65536 + (long)j * 256 + i] * inv);
}

// out = alpha*I - t
__global__ void k_ima(float alpha, const float* t, _Float16* out) {
  long idx = (long)blockIdx.x * 256 + threadIdx.x;    // 16*65536
  int j = (int)(idx & 255); int i = (int)((idx >> 8) & 255);
  float v = ((i == j) ? alpha : 0.f) - t[idx];
  out[idx] = (_Float16)v;
}

// depthwise conv residual (k=33 over seq) added to attn output, f16 out
__global__ void k_convres(const float* ao32, const _Float16* qkv, const float* cw,
                          _Float16* ao16) {
  long idx = (long)blockIdx.x * 256 + threadIdx.x;    // 2*16384*512
  int col = (int)(idx & 511); long r = idx >> 9;
  int n = (int)(r & 16383); int b = (int)(r >> 14);
  int h = col >> 6, d = col & 63;
  float s = 0.f;
  for (int tt = 0; tt < 33; ++tt) {
    int nn = n + tt - 16;
    if (nn >= 0 && nn < 16384)
      s += cw[h * 33 + tt] *
           (float)qkv[((long)(b * 16384 + nn)) * 1536 + 1024 + h * 64 + d];
  }
  ao16[idx] = (_Float16)(ao32[idx] + s);
}

__global__ void k_att2(const float* t1, const float* w2, const float* b2, float* alog) {
  long r = (long)blockIdx.x * 256 + threadIdx.x;      // 32768 rows
  const float* p = t1 + r * 128;
  float s = 0.f;
  for (int c = 0; c < 128; ++c) s += p[c] * w2[c];
  alog[r] = s + b2[0];
}

__global__ void k_softmax_seq(float* a, int n) {
  float* p = a + (long)blockIdx.x * n;
  __shared__ float red[256];
  int tid = threadIdx.x;
  float m = -1e30f;
  for (int c = tid; c < n; c += 256) m = fmaxf(m, p[c]);
  red[tid] = m; __syncthreads();
  for (int o = 128; o > 0; o >>= 1) { if (tid < o) red[tid] = fmaxf(red[tid], red[tid + o]); __syncthreads(); }
  m = red[0]; __syncthreads();
  float s = 0.f;
  for (int c = tid; c < n; c += 256) s += expf(p[c] - m);
  red[tid] = s; __syncthreads();
  for (int o = 128; o > 0; o >>= 1) { if (tid < o) red[tid] += red[tid + o]; __syncthreads(); }
  float inv = 1.f / red[0]; __syncthreads();
  for (int c = tid; c < n; c += 256) p[c] = expf(p[c] - m) * inv;
}

__global__ void k_pool(const float* a, const float* x, float* pooled) {
  int b = blockIdx.x; int d = threadIdx.x;            // block 512
  float s = 0.f;
  for (int n = 0; n < 16384; ++n)
    s += a[(long)b * 16384 + n] * x[((long)(b * 16384 + n)) * 512 + d];
  pooled[b * 512 + d] = s;
}

__global__ void k_head(const float* pooled, const float* w, const float* bb, float* out) {
  if (threadIdx.x == 0 && blockIdx.x == 0) {
    float lg[2][2];
    for (int b = 0; b < 2; ++b)
      for (int c = 0; c < 2; ++c) {
        float s = bb[c];
        for (int d = 0; d < 512; ++d) s += pooled[b * 512 + d] * w[d * 2 + c];
        lg[b][c] = s;
      }
    for (int b = 0; b < 2; ++b) { out[b * 2] = lg[b][0]; out[b * 2 + 1] = lg[b][1]; }
    for (int b = 0; b < 2; ++b) {
      float m = fmaxf(lg[b][0], lg[b][1]);
      float e0 = expf(lg[b][0] - m), e1 = expf(lg[b][1] - m);
      float s = e0 + e1;
      out[4 + b * 2] = e0 / s; out[5 + b * 2] = e1 / s;
    }
    for (int b = 0; b < 2; ++b) out[8 + b] = (lg[b][1] > lg[b][0]) ? 1.f : 0.f;
  }
}

// ---------------------------------------------------------------------------
// Host orchestration
// ---------------------------------------------------------------------------
extern "C" void kernel_launch(void* const* d_in, const int* in_sizes, int n_in,
                              void* d_out, int out_size, void* d_ws, size_t ws_size,
                              hipStream_t stream) {
  (void)in_sizes; (void)n_in; (void)out_size; (void)ws_size;
  const float* h      = (const float*)d_in[0];
  const float* fc1_w  = (const float*)d_in[1];
  const float* fc1_b  = (const float*)d_in[2];
  const float* ln_g   = (const float*)d_in[3];
  const float* ln_b   = (const float*)d_in[4];
  const float* qkv_w  = (const float*)d_in[5];
  const float* out_w  = (const float*)d_in[6];
  const float* out_b  = (const float*)d_in[7];
  const float* conv_w = (const float*)d_in[8];
  const float* att_w1 = (const float*)d_in[9];
  const float* att_b1 = (const float*)d_in[10];
  const float* att_w2 = (const float*)d_in[11];
  const float* att_b2 = (const float*)d_in[12];
  const float* fc2_w  = (const float*)d_in[13];
  const float* fc2_b  = (const float*)d_in[14];
  float* outp = (float*)d_out;

  const int R = 32768;  // b*n
  char* wp = (char*)d_ws;
  auto alloc = [&](size_t bytes) -> void* {
    void* p = (void*)wp; wp += (bytes + 255) & ~(size_t)255; return p;
  };
  float*    x32    = (float*)   alloc((size_t)R * 512 * 4);
  _Float16* x16    = (_Float16*)alloc((size_t)R * 512 * 2);
  _Float16* xn16   = (_Float16*)alloc((size_t)R * 512 * 2);
  _Float16* qkv16  = (_Float16*)alloc((size_t)R * 1536 * 2);
  _Float16* ql16   = (_Float16*)alloc((size_t)16 * 256 * 64 * 2);
  _Float16* kl16   = (_Float16*)alloc((size_t)16 * 256 * 64 * 2);
  float*    a2_32  = (float*)   alloc((size_t)16 * 65536 * 4);
  _Float16* x2_16  = (_Float16*)alloc((size_t)16 * 65536 * 2);
  float*    xz32   = (float*)   alloc((size_t)16 * 65536 * 4);
  _Float16* xz16   = (_Float16*)alloc((size_t)16 * 65536 * 2);
  float*    tmp32  = (float*)   alloc((size_t)16 * 65536 * 4);
  _Float16* aI16   = (_Float16*)alloc((size_t)16 * 65536 * 2);
  _Float16* za16   = (_Float16*)alloc((size_t)16 * 65536 * 2);
  _Float16* zb16   = (_Float16*)alloc((size_t)16 * 65536 * 2);
  _Float16* a3v16  = (_Float16*)alloc((size_t)16 * 256 * 64 * 2);
  _Float16* W16    = (_Float16*)alloc((size_t)16 * 256 * 64 * 2);
  _Float16* big16  = (_Float16*)alloc((size_t)16 * 16384 * 256 * 2); // s3/s1/h16
  float*    ao32   = (float*)   alloc((size_t)R * 512 * 4);
  _Float16* ao16   = (_Float16*)alloc((size_t)R * 512 * 2);
  float*    t132   = (float*)   alloc((size_t)R * 128 * 4);
  float*    alog   = (float*)   alloc((size_t)R * 4);
  float*    pooled = (float*)   alloc((size_t)2 * 512 * 4);
  float*    sums   = (float*)   alloc((size_t)4096 * 4);
  float*    scal   = (float*)   alloc((size_t)8 * 4);
  _Float16* fc1w16  = (_Float16*)alloc((size_t)1024 * 512 * 2);
  _Float16* qkvw16  = (_Float16*)alloc((size_t)2 * 512 * 1536 * 2);
  _Float16* outw16  = (_Float16*)alloc((size_t)2 * 512 * 512 * 2);
  _Float16* attw116 = (_Float16*)alloc((size_t)512 * 128 * 2);
  _Float16* h16 = big16;  // alias: h16 only live before the layers

  auto cvt = [&](const float* src, _Float16* dst, long n) {
    k_f2h<<<dim3((unsigned)((n + 255) / 256)), dim3(256), 0, stream>>>(src, dst, n);
  };
  cvt(h, h16, (long)R * 1024);
  cvt(fc1_w, fc1w16, 1024 * 512);
  cvt(qkv_w, qkvw16, 2 * 512 * 1536);
  cvt(out_w, outw16, 2 * 512 * 512);
  cvt(att_w1, attw116, 512 * 128);

  auto gemm = [&](const _Float16* A, const _Float16* Bm, float* C, _Float16* C16,
                  const float* bias, const float* resid, int M, int N, int K,
                  long lda, long ldb, long ldc,
                  long Asb, long Ash, long Bsb, long Bsh, long Csb, long Csh,
                  int Z, int H, int transB, int op, float alpha) {
    dim3 g((unsigned)(N / 64), (unsigned)(M / 256), (unsigned)Z);
    k_gemm<<<g, dim3(256), 0, stream>>>(A, Bm, C, C16, bias, resid, K, lda, ldb, ldc,
                                        Asb, Ash, Bsb, Bsh, Csb, Csh, H, transB, op, alpha);
  };

  // fc1: x = relu(h @ fc1_w + b)
  gemm(h16, fc1w16, x32, nullptr, fc1_b, nullptr, R, 512, 1024,
       1024, 512, 512, 0, 0, 0, 0, 0, 0, 1, 1, 0, 1, 1.f);

  const long SH = 16384;           // per-(b,h) stride of [bh,256,64] tensors
  const long SQ = (long)65536;     // per-(b,h) stride of [bh,256,256] tensors
  for (int l = 0; l < 2; ++l) {
    k_layernorm<<<dim3(R), dim3(256), 0, stream>>>(x32, ln_g + l * 512, ln_b + l * 512, xn16, 512);
    // qkv (f16 out only)
    gemm(xn16, qkvw16 + (size_t)l * 512 * 1536, nullptr, qkv16, nullptr, nullptr,
         R, 1536, 512, 512, 1536, 1536, 0, 0, 0, 0, 0, 0, 1, 1, 0, 0, 1.f);
    k_landmark<<<dim3(1024), dim3(256), 0, stream>>>(qkv16, ql16, 0);
    k_landmark<<<dim3(1024), dim3(256), 0, stream>>>(qkv16, kl16, 512);
    // attn2 scores -> x2_16 (alpha = 1/sqrt(64))
    gemm(ql16, kl16, nullptr, x2_16, nullptr, nullptr, 256, 256, 64,
         64, 64, 256, 8 * SH, SH, 8 * SH, SH, 8 * SQ, SQ, 16, 8, 1, 0, 0.125f);
    k_row_softmax<<<dim3(4096), dim3(256), 0, stream>>>(x2_16, a2_32, 256);
    // pinv init: z0 = x^T / (col*row)
    k_abssum_lastdim<<<dim3(16), dim3(256), 0, stream>>>(a2_32, sums);
    k_maxreduce<<<dim3(1), dim3(256), 0, stream>>>(sums, 4096, scal);
    k_abssum_dim2<<<dim3(16), dim3(256), 0, stream>>>(a2_32, sums);
    k_maxreduce<<<dim3(1), dim3(256), 0, stream>>>(sums, 4096, scal + 1);
    k_z0<<<dim3(4096), dim3(256), 0, stream>>>(a2_32, scal, scal + 1, za16);
    _Float16* zin = za16; _Float16* zout = zb16;
    for (int it = 0; it < 6; ++it) {
      gemm(x2_16, zin, xz32, xz16, nullptr, nullptr, 256, 256, 256,
           256, 256, 256, 8 * SQ, SQ, 8 * SQ, SQ, 8 * SQ, SQ, 16, 8, 0, 0, 1.f);
      k_ima<<<dim3(4096), dim3(256), 0, stream>>>(7.f, xz32, aI16);
      gemm(xz16, aI16, tmp32, nullptr, nullptr, nullptr, 256, 256, 256,
           256, 256, 256, 8 * SQ, SQ, 8 * SQ, SQ, 8 * SQ, SQ, 16, 8, 0, 0, 1.f);
      k_ima<<<dim3(4096), dim3(256), 0, stream>>>(15.f, tmp32, aI16);
      gemm(xz16, aI16, tmp32, nullptr, nullptr, nullptr, 256, 256, 256,
           256, 256, 256, 8 * SQ, SQ, 8 * SQ, SQ, 8 * SQ, SQ, 16, 8, 0, 0, 1.f);
      k_ima<<<dim3(4096), dim3(256), 0, stream>>>(13.f, tmp32, aI16);
      gemm(zin, aI16, nullptr, zout, nullptr, nullptr, 256, 256, 256,
           256, 256, 256, 8 * SQ, SQ, 8 * SQ, SQ, 8 * SQ, SQ, 16, 8, 0, 0, 0.25f);
      _Float16* tsw = zin; zin = zout; zout = tsw;
    }
    // attn3 scores: [bh,256,16384] -> big16
    gemm(ql16, qkv16 + 512, nullptr, big16, nullptr, nullptr, 256, 16384, 64,
         64, 1536, 16384, 8 * SH, SH, (long)16384 * 1536, 64,
         (long)8 * 256 * 16384, (long)256 * 16384, 16, 8, 1, 0, 0.125f);
    k_row_softmax<<<dim3(4096), dim3(256), 0, stream>>>(big16, nullptr, 16384);
    // attn3 @ v
    gemm(big16, qkv16 + 1024, nullptr, a3v16, nullptr, nullptr, 256, 64, 16384,
         16384, 1536, 64, (long)8 * 256 * 16384, (long)256 * 16384,
         (long)16384 * 1536, 64, 8 * SH, SH, 16, 8, 0, 0, 1.f);
    // W = pinv @ (attn3 @ v)
    gemm(zin, a3v16, nullptr, W16, nullptr, nullptr, 256, 64, 256,
         256, 64, 64, 8 * SQ, SQ, 8 * SH, SH, 8 * SH, SH, 16, 8, 0, 0, 1.f);
    // attn1 scores: [bh,16384,256] -> big16 (overwrites attn3 scores)
    gemm(qkv16, kl16, nullptr, big16, nullptr, nullptr, 16384, 256, 64,
         1536, 64, 256, (long)16384 * 1536, 64, 8 * SH, SH,
         (long)8 * 16384 * 256, (long)16384 * 256, 16, 8, 1, 0, 0.125f);
    k_row_softmax<<<dim3(262144), dim3(256), 0, stream>>>(big16, nullptr, 256);
    // out = attn1 @ W, scattered into [b,n,h*64+d]
    gemm(big16, W16, ao32, nullptr, nullptr, nullptr, 16384, 64, 256,
         256, 64, 512, (long)8 * 16384 * 256, (long)16384 * 256,
         8 * SH, SH, (long)16384 * 512, 64, 16, 8, 0, 0, 1.f);
    k_convres<<<dim3(65536), dim3(256), 0, stream>>>(ao32, qkv16, conv_w + l * 8 * 33, ao16);
    // out proj + bias + residual, in-place on x (also emit x16 for the tail)
    gemm(ao16, outw16 + (size_t)l * 512 * 512, x32, x16, out_b + l * 512, x32,
         R, 512, 512, 512, 512, 512, 0, 0, 0, 0, 0, 0, 1, 1, 0, 0, 1.f);
  }

  // attention-pooling tail
  gemm(x16, attw116, t132, nullptr, att_b1, nullptr, R, 128, 512,
       512, 128, 128, 0, 0, 0, 0, 0, 0, 1, 1, 0, 2, 1.f);   // tanh epilogue
  k_att2<<<dim3(128), dim3(256), 0, stream>>>(t132, att_w2, att_b2, alog);
  k_softmax_seq<<<dim3(2), dim3(256), 0, stream>>>(alog, 16384);
  k_pool<<<dim3(2), dim3(512), 0, stream>>>(alog, x32, pooled);
  k_head<<<dim3(1), dim3(32), 0, stream>>>(pooled, fc2_w, fc2_b, outp);
}
